// WindowAttention_53738630807643
// MI455X (gfx1250) — compile-verified, matrix operand order checked by reading, not run on hardware
//
#include <hip/hip_runtime.h>
#include <hip/hip_bf16.h>

// ---------------- CDNA5 types ----------------
typedef __attribute__((ext_vector_type(16))) __bf16 v16bf;
typedef __attribute__((ext_vector_type(8)))  float  v8f;
typedef __attribute__((ext_vector_type(4))) unsigned int v4u;
typedef __attribute__((ext_vector_type(8)))  int v8i;
typedef __attribute__((ext_vector_type(4)))  int v4i;

// ---------------- problem constants ----------------
#define WS        8
#define NTOK      64          // WS*WS tokens per window
#define DIM       128
#define HEADS     4
#define HD        32
#define SCALEQ    0.1767766952966369f   // 32^-0.5
#define NWIN_X    32
#define NWIN      (NWIN_X*NWIN_X)       // per batch image
#define HW        65536                 // 256*256
#define IMGW      256

// LDS layout (bytes). Padded leading dims chosen so every 16B fragment
// load is 16B-aligned and consecutive lanes hit distinct banks.
#define LDA       136   // bf16 elements; 272B rows (16B aligned, bank step 4)
#define LDVT      72    // bf16 elements; 144B rows
#define LDSS      68    // f32 elements;  272B rows
#define LDP       72    // bf16 elements; 144B rows

#define OFS_XA    0
#define OFS_Q     17408
#define OFS_K     34816
#define OFS_VT    52224
#define OFS_O     70656
#define OFS_S     88064
#define OFS_P     157696
#define OFS_XF32  OFS_S          // alias: xf32 dead before scores written
#define SMEM_BYTES 194560

static __device__ __forceinline__ unsigned short f2bf(float f) {
    unsigned u = __float_as_uint(f);
    u += 0x7FFFu + ((u >> 16) & 1u);      // round-to-nearest-even
    return (unsigned short)(u >> 16);
}
static __device__ __forceinline__ unsigned pack2bf(float lo, float hi) {
    return (unsigned)f2bf(lo) | ((unsigned)f2bf(hi) << 16);
}

union Frag {            // one WMMA bf16 A/B fragment: 16 bf16 per lane
    v16bf v;
    uint4 q[2];
};

static __device__ __forceinline__ void ldsFrag(Frag& f, const unsigned short* base,
                                               int row, int ld, int col) {
    const unsigned short* p = base + row * ld + col;
    f.q[0] = *(const uint4*)(p);
    f.q[1] = *(const uint4*)(p + 16);
}
static __device__ __forceinline__ void glbFrag(Frag& f, const unsigned short* base,
                                               int row, int ld, int col) {
    const unsigned short* p = base + row * ld + col;
    f.q[0] = *(const uint4*)(p);
    f.q[1] = *(const uint4*)(p + 16);
}

// ---------------- prep: bf16 weights + expanded bias table ----------------
__global__ __launch_bounds__(256) void wa_prep(
    const float* __restrict__ qkvw, const float* __restrict__ projw,
    const float* __restrict__ btable, const int* __restrict__ rindex,
    unsigned short* __restrict__ wq, unsigned short* __restrict__ wp,
    float* __restrict__ bbuf)
{
    int tid = blockIdx.x * 256 + threadIdx.x;
    if (tid < 3 * DIM * DIM) wq[tid] = f2bf(qkvw[tid]);          // 49152
    if (tid < DIM * DIM)     wp[tid] = f2bf(projw[tid]);         // 16384
    if (tid < HEADS * NTOK * NTOK) {                             // 16384
        int ij = tid >> 2, h = tid & 3;
        bbuf[h * (NTOK * NTOK) + ij] = btable[rindex[ij] * HEADS + h];
    }
}

// ---------------- fused window-attention kernel ----------------
__global__ __launch_bounds__(256, 1) void wa_main(
    const float* __restrict__ x,
    const unsigned short* __restrict__ wq,   // (384,128) bf16 row-major
    const float* __restrict__ qkvb,
    const unsigned short* __restrict__ wp,   // (128,128) bf16 row-major
    const float* __restrict__ projb,
    const float* __restrict__ bbuf,          // (4,64,64) f32
    float* __restrict__ out)
{
    extern __shared__ char smem[];
    unsigned short* xa = (unsigned short*)(smem + OFS_XA);
    unsigned short* qm = (unsigned short*)(smem + OFS_Q);
    unsigned short* km = (unsigned short*)(smem + OFS_K);
    unsigned short* vt = (unsigned short*)(smem + OFS_VT);
    unsigned short* om = (unsigned short*)(smem + OFS_O);
    float*          sm = (float*)(smem + OFS_S);
    unsigned short* pm = (unsigned short*)(smem + OFS_P);
    float*          xf = (float*)(smem + OFS_XF32);

    const int bid = blockIdx.x;
    const int bb  = bid >> 10;
    const int wh  = (bid & 1023) >> 5;
    const int ww  = bid & 31;
    const size_t xbase = (size_t)bb * DIM * HW + (size_t)(wh * WS) * IMGW + (size_t)(ww * WS);

    const int tid  = threadIdx.x;
    // wave index as a *scalar* so tile loops / region selects compile to SALU
    const int wvu  = __builtin_amdgcn_readfirstlane(tid >> 5);
    const int lane = tid & 31;
    const int l15  = lane & 15;
    const int khalf = (lane >> 4) << 3;   // 0 or 8

    // ---- stage 0/1: X tile -> LDS (token-major bf16) ----
#if __has_builtin(__builtin_amdgcn_tensor_load_to_lds)
    if (wvu == 0) {
        unsigned long long ga = (unsigned long long)(const void*)(x + xbase);
        v4u g0; g0.x = 1u;                         // count=1 valid descriptor
        g0.y = (unsigned)OFS_XF32;                 // lds_addr
        g0.z = (unsigned)(ga & 0xFFFFFFFFu);       // global_addr lo
        g0.w = (unsigned)((ga >> 32) & 0x1FFFFFFu) | (2u << 30);  // addr hi | type=2
        v8i g1;
        g1[0] = 0x00020000;                        // mask=0, data_size=4B
        g1[1] = (int)(256u << 16);                 // tensor_dim0 = 256 (low16 in [31:16])
        g1[2] = (int)(256u << 16);                 // tensor_dim1 = 256
        g1[3] = (int)(8u << 16);                   // tile_dim0 = 8
        g1[4] = (int)((128u << 16) | 8u);          // tile_dim1=8, tile_dim2=128
        g1[5] = 256;                               // tensor_dim0_stride = 256
        g1[6] = 0;                                 // stride0 hi16 | stride1 lo16
        g1[7] = 1;                                 // tensor_dim1_stride = 65536 >> 16
        v4i g2; g2[0] = 128; g2[1] = 0; g2[2] = 0; g2[3] = 0;   // tensor_dim2=128
        v4i g3; g3[0] = 0; g3[1] = 0; g3[2] = 0; g3[3] = 0;
        v8i g4; g4[0] = 0; g4[1] = 0; g4[2] = 0; g4[3] = 0;
                g4[4] = 0; g4[5] = 0; g4[6] = 0; g4[7] = 0;     // unused (count=1)
        __builtin_amdgcn_tensor_load_to_lds(g0, g1, g2, g3, g4, 0);
        __builtin_amdgcn_s_wait_tensorcnt(0);
    }
    __syncthreads();
    #pragma unroll
    for (int it = 0; it < 32; ++it) {               // transpose c-major f32 -> tok-major bf16
        int idx = tid + it * 256;
        int pos = idx & 63, c = idx >> 6;
        xa[pos * LDA + c] = f2bf(xf[c * 64 + pos]);
    }
#else
    #pragma unroll
    for (int it = 0; it < 4; ++it) {
        int s = tid + it * 256;
        int c = s >> 3, r = s & 7;
        const float4* gp = (const float4*)(x + xbase + (size_t)c * HW + (size_t)r * IMGW);
        float4 v0 = gp[0], v1 = gp[1];
        int p0 = (r * 8) * LDA + c;
        xa[p0          ] = f2bf(v0.x); xa[p0 + LDA    ] = f2bf(v0.y);
        xa[p0 + 2 * LDA] = f2bf(v0.z); xa[p0 + 3 * LDA] = f2bf(v0.w);
        xa[p0 + 4 * LDA] = f2bf(v1.x); xa[p0 + 5 * LDA] = f2bf(v1.y);
        xa[p0 + 6 * LDA] = f2bf(v1.z); xa[p0 + 7 * LDA] = f2bf(v1.w);
    }
#endif
    __syncthreads();

    // ---- stage 2: QKV GEMM 64x384x128 ----
    // One wave owns a 64x16 column strip (one nt, all 4 mt). Weight fragments
    // loaded once per strip; A fragments batch-loaded per kb so WMMAs issue
    // back-to-back behind a single dscnt wait.
    for (int nt = wvu; nt < 24; nt += 8) {            // nt scalar
        int nB = nt * 16 + l15;                       // per-lane
        Frag b[4];
        #pragma unroll
        for (int kb = 0; kb < 4; ++kb)
            glbFrag(b[kb], wq, nB, DIM, kb * 32 + khalf);
        v8f acc[4];
        #pragma unroll
        for (int mt = 0; mt < 4; ++mt)
            #pragma unroll
            for (int g = 0; g < 8; ++g) acc[mt][g] = 0.f;
        #pragma unroll
        for (int kb = 0; kb < 4; ++kb) {
            Frag a[4];
            #pragma unroll
            for (int mt = 0; mt < 4; ++mt)
                ldsFrag(a[mt], xa, mt * 16 + l15, LDA, kb * 32 + khalf);
            #pragma unroll
            for (int mt = 0; mt < 4; ++mt)
                acc[mt] = __builtin_amdgcn_wmma_f32_16x16x32_bf16(false, a[mt].v, false, b[kb].v,
                                                                  (short)0, acc[mt], false, false);
        }
        float bias = qkvb[nB];
        int region = nt >> 3;                         // scalar: 0:q 1:k 2:v
        if (region == 0) {
            #pragma unroll
            for (int mt = 0; mt < 4; ++mt)
                #pragma unroll
                for (int g = 0; g < 8; ++g)
                    qm[(mt * 16 + g + khalf) * LDA + nB] = f2bf((acc[mt][g] + bias) * SCALEQ);
        } else if (region == 1) {
            #pragma unroll
            for (int mt = 0; mt < 4; ++mt)
                #pragma unroll
                for (int g = 0; g < 8; ++g)
                    km[(mt * 16 + g + khalf) * LDA + (nB - 128)] = f2bf(acc[mt][g] + bias);
        } else {
            #pragma unroll
            for (int mt = 0; mt < 4; ++mt)
                #pragma unroll
                for (int g = 0; g < 8; ++g)
                    vt[(nB - 256) * LDVT + mt * 16 + g + khalf] = f2bf(acc[mt][g] + bias);
        }
    }
    __syncthreads();

    // ---- stage 3: scores = Q K^T + bias ----
    // One wave owns an (h, mt) pair: Q fragment loaded once; 4 K fragments
    // batch-loaded, then 4 WMMAs back-to-back.
    for (int p = wvu; p < 16; p += 8) {               // scalar
        int h = p >> 2, mt = p & 3;
        int m0 = mt * 16;
        Frag a, b[4];
        ldsFrag(a, qm, m0 + l15, LDA, h * HD + khalf);
        #pragma unroll
        for (int nt = 0; nt < 4; ++nt)
            ldsFrag(b[nt], km, nt * 16 + l15, LDA, h * HD + khalf);
        v8f acc[4];
        #pragma unroll
        for (int nt = 0; nt < 4; ++nt) {
            #pragma unroll
            for (int g = 0; g < 8; ++g) acc[nt][g] = 0.f;
            acc[nt] = __builtin_amdgcn_wmma_f32_16x16x32_bf16(false, a.v, false, b[nt].v,
                                                              (short)0, acc[nt], false, false);
        }
        #pragma unroll
        for (int nt = 0; nt < 4; ++nt) {
            int nfix = nt * 16 + l15;
            #pragma unroll
            for (int g = 0; g < 8; ++g) {
                int m = m0 + g + khalf;
                sm[(h * NTOK + m) * LDSS + nfix] = acc[nt][g] + bbuf[(h * NTOK + m) * NTOK + nfix];
            }
        }
    }
    __syncthreads();

    // ---- stage 4: softmax, one (head,row) per thread; two-pass, no e[] array ----
    {
        int h = tid >> 6, i = tid & 63;
        const float4* rs4 = (const float4*)(sm + (h * NTOK + i) * LDSS);
        float mx = -3.4e38f;
        #pragma unroll
        for (int j = 0; j < 16; ++j) {
            float4 v = rs4[j];
            mx = fmaxf(mx, fmaxf(fmaxf(v.x, v.y), fmaxf(v.z, v.w)));
        }
        float sum = 0.f;
        #pragma unroll
        for (int j = 0; j < 16; ++j) {
            float4 v = rs4[j];
            sum += __expf(v.x - mx) + __expf(v.y - mx)
                 + __expf(v.z - mx) + __expf(v.w - mx);
        }
        float inv = __frcp_rn(sum);
        unsigned* pr = (unsigned*)(pm + (h * NTOK + i) * LDP);
        #pragma unroll
        for (int j = 0; j < 16; ++j) {
            float4 v = rs4[j];
            pr[2 * j]     = pack2bf(__expf(v.x - mx) * inv, __expf(v.y - mx) * inv);
            pr[2 * j + 1] = pack2bf(__expf(v.z - mx) * inv, __expf(v.w - mx) * inv);
        }
    }
    __syncthreads();

    // ---- stage 5: out = P V ----
    // One wave owns an (h, mt) pair: all fragments batch-loaded, 4 WMMAs.
    for (int p = wvu; p < 16; p += 8) {               // scalar
        int h = p >> 2, mt = p & 3;
        int m0 = mt * 16;
        Frag a[2], b[4];
        #pragma unroll
        for (int kb = 0; kb < 2; ++kb) {
            ldsFrag(a[kb], pm, h * NTOK + m0 + l15, LDP, kb * 32 + khalf);
            #pragma unroll
            for (int nt = 0; nt < 2; ++nt)
                ldsFrag(b[kb * 2 + nt], vt, h * HD + nt * 16 + l15, LDVT, kb * 32 + khalf);
        }
        v8f acc[2];
        #pragma unroll
        for (int nt = 0; nt < 2; ++nt)
            #pragma unroll
            for (int g = 0; g < 8; ++g) acc[nt][g] = 0.f;
        #pragma unroll
        for (int kb = 0; kb < 2; ++kb)
            #pragma unroll
            for (int nt = 0; nt < 2; ++nt)
                acc[nt] = __builtin_amdgcn_wmma_f32_16x16x32_bf16(false, a[kb].v, false, b[kb * 2 + nt].v,
                                                                  (short)0, acc[nt], false, false);
        #pragma unroll
        for (int nt = 0; nt < 2; ++nt) {
            int och = h * HD + nt * 16 + l15;
            #pragma unroll
            for (int g = 0; g < 8; ++g)
                om[(m0 + g + khalf) * LDA + och] = f2bf(acc[nt][g]);
        }
    }
    __syncthreads();

    // ---- stage 6: proj GEMM 64x128x128 + bias, store bchw f32 ----
    // One wave owns one nt column strip (all 4 mt): proj_w fragments loaded once.
    {
        int nB = wvu * 16 + l15;
        Frag b[4];
        #pragma unroll
        for (int kb = 0; kb < 4; ++kb)
            glbFrag(b[kb], wp, nB, DIM, kb * 32 + khalf);
        v8f acc[4];
        #pragma unroll
        for (int mt = 0; mt < 4; ++mt)
            #pragma unroll
            for (int g = 0; g < 8; ++g) acc[mt][g] = 0.f;
        #pragma unroll
        for (int kb = 0; kb < 4; ++kb) {
            Frag a[4];
            #pragma unroll
            for (int mt = 0; mt < 4; ++mt)
                ldsFrag(a[mt], om, mt * 16 + l15, LDA, kb * 32 + khalf);
            #pragma unroll
            for (int mt = 0; mt < 4; ++mt)
                acc[mt] = __builtin_amdgcn_wmma_f32_16x16x32_bf16(false, a[mt].v, false, b[kb].v,
                                                                  (short)0, acc[mt], false, false);
        }
        float pb = projb[nB];
        float* obase = out + ((size_t)bb * DIM + (size_t)nB) * HW
                           + (size_t)(wh * WS) * IMGW + (size_t)(ww * WS);
        #pragma unroll
        for (int mt = 0; mt < 4; ++mt) {
            #pragma unroll
            for (int g = 0; g < 8; ++g) {
                int m = mt * 16 + g + khalf;
                int r = m >> 3, cc = m & 7;
                obase[(size_t)r * IMGW + cc] = acc[mt][g] + pb;
            }
        }
    }
}

// ---------------- host launch ----------------
extern "C" void kernel_launch(void* const* d_in, const int* in_sizes, int n_in,
                              void* d_out, int out_size, void* d_ws, size_t ws_size,
                              hipStream_t stream) {
    const float* x      = (const float*)d_in[0];
    const float* qkv_w  = (const float*)d_in[1];
    const float* qkv_b  = (const float*)d_in[2];
    const float* proj_w = (const float*)d_in[3];
    const float* proj_b = (const float*)d_in[4];
    const float* btable = (const float*)d_in[5];
    const int*   rindex = (const int*)d_in[6];
    float* out = (float*)d_out;

    char* ws = (char*)d_ws;
    unsigned short* wq   = (unsigned short*)(ws);            // 98304 B
    unsigned short* wp   = (unsigned short*)(ws + 98304);    // 32768 B
    float*          bbuf = (float*)(ws + 131072);            // 65536 B

    wa_prep<<<192, 256, 0, stream>>>(qkv_w, proj_w, btable, rindex, wq, wp, bbuf);

    int nblocks = 8 * NWIN;   // 8192 windows
    wa_main<<<nblocks, 256, SMEM_BYTES, stream>>>(x, wq, qkv_b, wp, proj_b, bbuf, out);
}
